// OnDemandLinear_5944234738232
// MI455X (gfx1250) — compile-verified, matrix operand order checked by reading, not run on hardware
//
#include <hip/hip_runtime.h>

typedef __attribute__((ext_vector_type(16))) _Float16 v16h;
typedef __attribute__((ext_vector_type(8)))  _Float16 v8h;
typedef __attribute__((ext_vector_type(8)))  float    v8f;
typedef __attribute__((ext_vector_type(4)))  int      v4i;

#define M_DIM 8192      // B*S = 4*2048
#define N_DIM 16384     // OUT
#define K_DIM 4096      // IN
#define BM 128
#define BN 128
#define BK 32
#define LSTR 40         // LDS row stride in halves (BK + 8 pad -> conflict-free b128)
#define KTILES (K_DIM / BK)

#define AS1 __attribute__((address_space(1)))
#define AS3 __attribute__((address_space(3)))

// Async global->LDS path (CDNA5 GLOBAL_LOAD_ASYNC_TO_LDS_*, ASYNCcnt-tracked)
#if defined(__gfx1250__) && __has_builtin(__builtin_amdgcn_global_load_async_to_lds_b128)
#define USE_ASYNC_LDS 1
#else
#define USE_ASYNC_LDS 0
#endif

static __device__ __forceinline__ void wait_async0() {
#if defined(__gfx1250__)
#if __has_builtin(__builtin_amdgcn_s_wait_asynccnt)
    __builtin_amdgcn_s_wait_asynccnt(0);
#else
    asm volatile("s_wait_asynccnt 0x0" ::: "memory");
#endif
#endif
}

static constexpr float DQ_SCALE = 0.015625f;   // 2^-6, exact in fp16

__global__ __launch_bounds__(256) void dq_linear_wmma(
    const _Float16* __restrict__ X,
    const int*      __restrict__ W,
    const _Float16* __restrict__ bias,
    _Float16*       __restrict__ out)
{
    __shared__ _Float16 sA[2][BM * LSTR];
    __shared__ _Float16 sB[2][BN * LSTR];

    const int tid  = threadIdx.x;
    const int lane = tid & 31;
    const int wave = tid >> 5;
    const int lr   = lane & 15;        // row/col within 16x16 fragment
    const int lh   = lane >> 4;        // lane-half selects K-chunks / M-half
    const int wm   = (wave & 3) * 32;  // wave M offset in block
    const int wn   = (wave >> 2) * 64; // wave N offset in block

    // M is the FAST grid dimension: the ~64 consecutively-scheduled M-blocks
    // share one 2MB weight strip (hot in L2), and the whole 64MB activation
    // matrix fits in the 192MB L2 across N-strips -> weights stream from HBM
    // approximately once (256MB) instead of once per M-sweep (16GB).
    const int m0 = blockIdx.x * BM;
    const int n0 = blockIdx.y * BN;

    // ---- global staging mapping: 256 threads cover 128 rows x 32 K ----
    const int row = tid >> 1;          // 0..127
    const int kof = (tid & 1) * 16;    // 0 or 16 (halves / ints)
    const _Float16* gA = X + (size_t)(m0 + row) * K_DIM + kof;
    const int*      gW = W + (size_t)(n0 + row) * K_DIM + kof;

#if !USE_ASYNC_LDS
    v8h ra0, ra1;          // fallback: A staged through VGPRs
#endif
    v8h rb0, rb1;          // 16 dequantized weight halves

    // A tile: straight fp16 copy -> async DMA into LDS (no VGPR staging)
    auto stage_A = [&](int kt, int buf) {
        const _Float16* src = gA + kt * BK;
        _Float16* dst = &sA[buf][row * LSTR + kof];
#if USE_ASYNC_LDS
        __builtin_amdgcn_global_load_async_to_lds_b128(
            (AS1 v4i*)src, (AS3 v4i*)dst, 0, 0);
        __builtin_amdgcn_global_load_async_to_lds_b128(
            (AS1 v4i*)(src + 8), (AS3 v4i*)(dst + 8), 0, 0);
#else
        (void)dst;
        ra0 = *(const v8h*)(src);
        ra1 = *(const v8h*)(src + 8);
#endif
    };

    // B tile: int32 -> fp16 dequant through VGPRs
    auto load_B = [&](int kt) {
        const int* pw = gW + kt * BK;
        v4i w0 = *(const v4i*)(pw);
        v4i w1 = *(const v4i*)(pw + 4);
        v4i w2 = *(const v4i*)(pw + 8);
        v4i w3 = *(const v4i*)(pw + 12);
#pragma unroll
        for (int i = 0; i < 4; ++i) {
            rb0[i]     = (_Float16)((float)w0[i] * DQ_SCALE);
            rb0[i + 4] = (_Float16)((float)w1[i] * DQ_SCALE);
            rb1[i]     = (_Float16)((float)w2[i] * DQ_SCALE);
            rb1[i + 4] = (_Float16)((float)w3[i] * DQ_SCALE);
        }
    };

    auto store_lds = [&](int buf) {
        _Float16* db = &sB[buf][row * LSTR + kof];
        *(v8h*)(db)     = rb0;
        *(v8h*)(db + 8) = rb1;
#if !USE_ASYNC_LDS
        _Float16* da = &sA[buf][row * LSTR + kof];
        *(v8h*)(da)     = ra0;
        *(v8h*)(da + 8) = ra1;
#endif
    };

    // ISA 16-bit fragment layout: lane holds row (A) / col (B) = lr;
    // lane-half lh picks K-chunks {lh*8..lh*8+7} and {16+lh*8..16+lh*8+7}.
    auto load_frag = [&](const _Float16* base) -> v16h {
        v8h lo = *(const v8h*)(base + lh * 8);
        v8h hi = *(const v8h*)(base + 16 + lh * 8);
        return __builtin_shufflevector(lo, hi,
            0, 1, 2, 3, 4, 5, 6, 7, 8, 9, 10, 11, 12, 13, 14, 15);
    };

    v8f acc[2][4];
#pragma unroll
    for (int mi = 0; mi < 2; ++mi)
#pragma unroll
        for (int ni = 0; ni < 4; ++ni)
#pragma unroll
            for (int v = 0; v < 8; ++v)
                acc[mi][ni][v] = 0.0f;

    auto compute = [&](int buf) {
        v16h afr[2], bfr[4];
#pragma unroll
        for (int mi = 0; mi < 2; ++mi)
            afr[mi] = load_frag(&sA[buf][(wm + mi * 16 + lr) * LSTR]);
#pragma unroll
        for (int ni = 0; ni < 4; ++ni)
            bfr[ni] = load_frag(&sB[buf][(wn + ni * 16 + lr) * LSTR]);
#pragma unroll
        for (int mi = 0; mi < 2; ++mi)
#pragma unroll
            for (int ni = 0; ni < 4; ++ni)
                acc[mi][ni] = __builtin_amdgcn_wmma_f32_16x16x32_f16(
                    /*neg_a=*/false, afr[mi], /*neg_b=*/false, bfr[ni],
                    /*c_mod=*/(short)0, acc[mi][ni],
                    /*reuse_a=*/false, /*reuse_b=*/false);
    };

    // ---- software-pipelined main loop (double-buffered LDS) ----
    stage_A(0, 0);
    load_B(0);
    store_lds(0);
    wait_async0();
    __syncthreads();

    int buf = 0;
#pragma unroll 1
    for (int kt = 0; kt < KTILES; ++kt) {
        if (kt + 1 < KTILES) {
            stage_A(kt + 1, buf ^ 1);      // async DMA straight into LDS
            load_B(kt + 1);                // overlap dequant loads with WMMAs
            if (kt + 2 < KTILES)
                __builtin_prefetch(gW + (kt + 2) * BK, 0, 0);
        }
        compute(buf);
        if (kt + 1 < KTILES) {
            store_lds(buf ^ 1);
            wait_async0();                 // A(kt+1) resident in LDS
            __syncthreads();
        }
        buf ^= 1;
    }

    // ---- epilogue: acc(f32) + bias -> fp16 ----
    // C/D layout: lane = (m>>3)*16 + n; VGPR v = m&7.
#pragma unroll
    for (int ni = 0; ni < 4; ++ni) {
        const int n = n0 + wn + ni * 16 + lr;
        const float bv = (float)bias[n];
#pragma unroll
        for (int mi = 0; mi < 2; ++mi) {
#pragma unroll
            for (int v = 0; v < 8; ++v) {
                const int m = m0 + wm + mi * 16 + lh * 8 + v;
                out[(size_t)m * N_DIM + n] = (_Float16)(acc[mi][ni][v] + bv);
            }
        }
    }
}

extern "C" void kernel_launch(void* const* d_in, const int* in_sizes, int n_in,
                              void* d_out, int out_size, void* d_ws, size_t ws_size,
                              hipStream_t stream) {
    const _Float16* X    = (const _Float16*)d_in[0];
    const int*      W    = (const int*)d_in[1];
    const _Float16* bias = (const _Float16*)d_in[2];
    _Float16*       out  = (_Float16*)d_out;

    dim3 grid(M_DIM / BM, N_DIM / BN);   // 64 (M, fast) x 128 (N, slow)
    dq_linear_wmma<<<grid, 256, 0, stream>>>(X, W, bias, out);
}
